// SelfAttention_45956150068010
// MI455X (gfx1250) — matrix operand order repaired and checked by reading, not compiled, not run on hardware
//
#include <hip/hip_runtime.h>
#include <hip/hip_bf16.h>

// ---------------------------------------------------------------------------
// Self-attention for MI455X (gfx1250): bf16 WMMA, f32 accumulate, TDM-fed
// flash attention.
//   B=4, S=2048, E=1024, H=16, D=64
// Pipeline:
//   1) one-shot f32->bf16 conversion of activations + weights (HBM-bound)
//   2) Q/K/V projections as bf16 WMMA GEMMs (V stored head-transposed)
//   3) flash attention: K/V tiles DMA'd to LDS by the Tensor Data Mover
//      (double-buffered, s_wait_tensorcnt), online softmax via DPP16 rows
//   4) output projection GEMM -> f32
// ---------------------------------------------------------------------------

#define BATCH    4
#define SEQ      2048
#define EMBED    1024
#define HEADS    16
#define HEAD_DIM 64

typedef __attribute__((ext_vector_type(16))) __bf16 v16bf;
typedef __attribute__((ext_vector_type(8)))  __bf16 v8bf;
typedef __attribute__((ext_vector_type(8)))  float  v8f;
typedef __attribute__((ext_vector_type(4)))  unsigned int u32x4;
typedef __attribute__((ext_vector_type(8)))  int i32x8;
typedef __attribute__((ext_vector_type(4)))  int i32x4;

static __device__ inline v8f zero8() {
    v8f z = {0.f,0.f,0.f,0.f,0.f,0.f,0.f,0.f};
    return z;
}

static __device__ inline v16bf cat8(v8bf lo, v8bf hi) {
    v16bf r;
#pragma unroll
    for (int i = 0; i < 8; ++i) { r[i] = lo[i]; r[8 + i] = hi[i]; }
    return r;
}

// ---- DPP16 in-row butterfly reductions (wave32: rows = lanes 0-15 / 16-31) --
#define DPP_QUAD_XOR1   0xB1   // quad_perm(1,0,3,2)
#define DPP_QUAD_XOR2   0x4E   // quad_perm(2,3,0,1)
#define DPP_ROW_HMIRROR 0x141  // row_half_mirror
#define DPP_ROW_MIRROR  0x140  // row_mirror

template <int CTRL>
static __device__ inline float dpp_mov(float x) {
    int xi = __builtin_bit_cast(int, x);
    int r  = __builtin_amdgcn_update_dpp(0, xi, CTRL, 0xf, 0xf, true);
    return __builtin_bit_cast(float, r);
}

static __device__ inline float row_max16(float x) {
    x = fmaxf(x, dpp_mov<DPP_QUAD_XOR1>(x));
    x = fmaxf(x, dpp_mov<DPP_QUAD_XOR2>(x));
    x = fmaxf(x, dpp_mov<DPP_ROW_HMIRROR>(x));
    x = fmaxf(x, dpp_mov<DPP_ROW_MIRROR>(x));
    return x;
}

static __device__ inline float row_sum16(float x) {
    x += dpp_mov<DPP_QUAD_XOR1>(x);
    x += dpp_mov<DPP_QUAD_XOR2>(x);
    x += dpp_mov<DPP_ROW_HMIRROR>(x);
    x += dpp_mov<DPP_ROW_MIRROR>(x);
    return x;
}

// ---------------------------------------------------------------------------
// Tensor Data Mover: 2D bf16 tile load (global, strided) -> LDS (padded rows).
// D# per CDNA5 ISA. Groups 2/3 zero => 2D tensor.
// pad_interval_code: pad after 2^(c+1) DWORDs; pad_amount_code: (c+1) DWORDs.
// 6-arg builtin form (clang-23 / therock-10.0 headers).
// ---------------------------------------------------------------------------
static __device__ inline void tdm_load_2d_bf16(
    unsigned lds_off, const void* gaddr,
    unsigned tile_d0, unsigned tile_d1,
    unsigned tensor_d0, unsigned tensor_d1, unsigned stride_elems,
    unsigned pad_interval_code, unsigned pad_amount_code)
{
    const unsigned long long ga = (unsigned long long)(size_t)gaddr;
    u32x4 g0;
    g0[0] = 1u;                                   // count=1, user descriptor
    g0[1] = lds_off;                              // LDS byte address
    g0[2] = (unsigned)ga;                         // global_addr[31:0]
    g0[3] = (unsigned)((ga >> 32) & 0x01FFFFFFu)  // global_addr[56:32]
          | (2u << 30);                           // type=2 ("image")
    i32x8 g1;
    g1[0] = (int)((1u << 16)                      // data_size: 2 bytes
                | (1u << 20)                      // pad_enable
                | (pad_interval_code << 22)
                | (pad_amount_code << 25));
    g1[1] = (int)((tensor_d0 & 0xFFFFu) << 16);   // barrier_addr=0 | dim0 lo16
    g1[2] = (int)((tensor_d0 >> 16) | ((tensor_d1 & 0xFFFFu) << 16));
    g1[3] = (int)((tensor_d1 >> 16) | (tile_d0 << 16));
    g1[4] = (int)(tile_d1 & 0xFFFFu);             // tile_dim1 | tile_dim2=0
    g1[5] = (int)stride_elems;                    // tensor_dim0_stride[31:0]
    g1[6] = 0;                                    // stride hi | dim1_stride lo
    g1[7] = 0;
    const i32x4 gz4 = {0, 0, 0, 0};
    const i32x8 gz8 = {0, 0, 0, 0, 0, 0, 0, 0};
    __builtin_amdgcn_tensor_load_to_lds(g0, g1, gz4, gz4, gz8, 0);
}

// ---------------------------------------------------------------------------
// Elementwise f32 -> bf16 conversion (8 elements / thread).
// ---------------------------------------------------------------------------
__global__ __launch_bounds__(256) void cvt_bf16_kernel(
    const float* __restrict__ in, unsigned short* __restrict__ out, int n8)
{
    const int t = blockIdx.x * 256 + threadIdx.x;
    if (t >= n8) return;
    const size_t i = (size_t)t * 8;
    v8f x = *(const v8f*)(in + i);
    v8bf y;
#pragma unroll
    for (int j = 0; j < 8; ++j) y[j] = (__bf16)x[j];
    *(v8bf*)((__bf16*)out + i) = y;
}

// ---------------------------------------------------------------------------
// GEMM: C[M][N] = A[M][K] * W[N][K]^T + bias, bf16 in / f32 accumulate.
// MODE 0: bf16 row-major out; MODE 1: f32 row-major out;
// MODE 2: bf16 out scattered to [b][h][d][s] (head-transposed V).
// Block = 128 threads (4 waves); wave tile 64x64; block tile 128x128.
// ---------------------------------------------------------------------------
template <int MODE>
__global__ __launch_bounds__(128) void gemm_bias_kernel(
    const unsigned short* __restrict__ Au, const unsigned short* __restrict__ Wu,
    const float* __restrict__ bias, void* __restrict__ Cout,
    int M, int N, int Kd, int Srows)
{
    const __bf16* A = (const __bf16*)Au;
    const __bf16* W = (const __bf16*)Wu;

    const int lane = threadIdx.x & 31;
    const int wave = threadIdx.x >> 5;
    const int nl   = lane & 15;
    const int half = lane >> 4;
    const int mbase = blockIdx.y * 128 + (wave >> 1) * 64;
    const int nbase = blockIdx.x * 128 + (wave & 1) * 64;

    v8f acc[4][4];
#pragma unroll
    for (int mi = 0; mi < 4; ++mi)
#pragma unroll
        for (int ni = 0; ni < 4; ++ni) acc[mi][ni] = zero8();

    for (int kb = 0; kb < Kd; kb += 32) {
        v16bf af[4], bf_[4];
#pragma unroll
        for (int mi = 0; mi < 4; ++mi) {
            const __bf16* p = A + (size_t)(mbase + mi * 16 + nl) * Kd + kb + half * 8;
            af[mi] = cat8(*(const v8bf*)p, *(const v8bf*)(p + 16));
        }
#pragma unroll
        for (int ni = 0; ni < 4; ++ni) {
            const __bf16* p = W + (size_t)(nbase + ni * 16 + nl) * Kd + kb + half * 8;
            bf_[ni] = cat8(*(const v8bf*)p, *(const v8bf*)(p + 16));
        }
#pragma unroll
        for (int mi = 0; mi < 4; ++mi)
#pragma unroll
            for (int ni = 0; ni < 4; ++ni)
                acc[mi][ni] = __builtin_amdgcn_wmma_f32_16x16x32_bf16(
                    false, af[mi], false, bf_[ni], (short)0, acc[mi][ni], false, false);
    }

    // Epilogue: D fragment row m = r + 8*half, col n = nl.
#pragma unroll
    for (int mi = 0; mi < 4; ++mi) {
#pragma unroll
        for (int ni = 0; ni < 4; ++ni) {
            const int col = nbase + ni * 16 + nl;
            const float bi = bias[col];
#pragma unroll
            for (int r = 0; r < 8; ++r) {
                const int row = mbase + mi * 16 + r + 8 * half;
                const float v = acc[mi][ni][r] + bi;
                if constexpr (MODE == 0) {
                    ((__bf16*)Cout)[(size_t)row * N + col] = (__bf16)v;
                } else if constexpr (MODE == 1) {
                    ((float*)Cout)[(size_t)row * N + col] = v;
                } else {
                    const int b = row / Srows, s = row % Srows;
                    const int h = col >> 6,   d = col & 63;
                    ((__bf16*)Cout)[(((size_t)b * HEADS + h) * HEAD_DIM + d) * Srows + s] = (__bf16)v;
                }
            }
        }
    }
}

// ---------------------------------------------------------------------------
// Flash attention: grid (B*H, S/64), 128 threads (4 waves).
// Wave w owns query rows [q0, q0+16). K/V tiles for the whole block are
// double-buffered in LDS by the Tensor Data Mover (issued by wave 0):
//   K tile: 32 keys x 64 d, LDS row stride 72 bf16 (144 B, pad 16 B)
//   V tile: 64 d x 32 keys, LDS row stride 40 bf16 ( 80 B, pad 16 B)
// ---------------------------------------------------------------------------
#define KROW 72
#define VROW 40

__global__ __launch_bounds__(128) void attention_kernel(
    const unsigned short* __restrict__ Qu, const unsigned short* __restrict__ Ku,
    const unsigned short* __restrict__ Vtu, unsigned short* __restrict__ Ou, int S)
{
    const __bf16* Q  = (const __bf16*)Qu;
    const __bf16* Kc = (const __bf16*)Ku;
    const __bf16* Vt = (const __bf16*)Vtu;
    __bf16*       O  = (__bf16*)Ou;

    const int lane = threadIdx.x & 31;
    const int wave = threadIdx.x >> 5;
    const int nl   = lane & 15;
    const int half = lane >> 4;
    const int bh = blockIdx.x;
    const int b  = bh >> 4;        // H == 16
    const int h  = bh & 15;
    const int q0 = blockIdx.y * 64 + wave * 16;

    __shared__ __attribute__((aligned(16))) __bf16 kbuf[2][32 * KROW];
    __shared__ __attribute__((aligned(16))) __bf16 vbuf[2][64 * VROW];
    __shared__ __attribute__((aligned(16))) __bf16 pbuf[4][16][40];

    const __bf16* kglob = Kc + ((size_t)(b * S)) * EMBED + h * HEAD_DIM;
    const __bf16* vglob = Vt + ((size_t)(b * HEADS + h)) * HEAD_DIM * S;

    // Prologue: TDM-load tile 0 into buffer 0 (wave 0 only; EXEC-independent).
    if (wave == 0) {
        tdm_load_2d_bf16((unsigned)(size_t)&kbuf[0][0], kglob,
                         64, 32, EMBED, S, EMBED, 4, 3);
        tdm_load_2d_bf16((unsigned)(size_t)&vbuf[0][0], vglob,
                         32, 64, S, HEAD_DIM, S, 3, 3);
    }

    // Q A-fragments for d in [0,32) and [32,64).
    const __bf16* qp = Q + ((size_t)(b * S + q0 + nl)) * EMBED + h * HEAD_DIM;
    const v16bf qa0 = cat8(*(const v8bf*)(qp + half * 8),      *(const v8bf*)(qp + 16 + half * 8));
    const v16bf qa1 = cat8(*(const v8bf*)(qp + 32 + half * 8), *(const v8bf*)(qp + 48 + half * 8));

    v8f o[4];
#pragma unroll
    for (int j = 0; j < 4; ++j) o[j] = zero8();
    float rmax[8], rsum[8], corr[8];
#pragma unroll
    for (int r = 0; r < 8; ++r) { rmax[r] = -1e30f; rsum[r] = 0.f; }

    int ib = 0;
    for (int kt = 0; kt < S; kt += 32, ib ^= 1) {
        // Issue next tile pair; wait for current pair; release the block.
        if (wave == 0) {
            const int nkt = kt + 32;
            if (nkt < S) {
                tdm_load_2d_bf16((unsigned)(size_t)&kbuf[ib ^ 1][0],
                                 kglob + (size_t)nkt * EMBED,
                                 64, 32, EMBED, S, EMBED, 4, 3);
                tdm_load_2d_bf16((unsigned)(size_t)&vbuf[ib ^ 1][0],
                                 vglob + nkt,
                                 32, 64, S, HEAD_DIM, S, 3, 3);
                __builtin_amdgcn_s_wait_tensorcnt((short)2);
            } else {
                __builtin_amdgcn_s_wait_tensorcnt((short)0);
            }
        }
        __syncthreads();

        const __bf16* klds = &kbuf[ib][0];
        const __bf16* vlds = &vbuf[ib][0];

        // S = Q * K^T for 32 keys (2 column fragments x 2 k-steps over D=64).
        v8f sf[2];
#pragma unroll
        for (int j = 0; j < 2; ++j) {
            sf[j] = zero8();
            const __bf16* kp = klds + (j * 16 + nl) * KROW;
            const v16bf kb0 = cat8(*(const v8bf*)(kp + half * 8),      *(const v8bf*)(kp + 16 + half * 8));
            const v16bf kb1 = cat8(*(const v8bf*)(kp + 32 + half * 8), *(const v8bf*)(kp + 48 + half * 8));
            sf[j] = __builtin_amdgcn_wmma_f32_16x16x32_bf16(false, qa0, false, kb0, (short)0, sf[j], false, false);
            sf[j] = __builtin_amdgcn_wmma_f32_16x16x32_bf16(false, qa1, false, kb1, (short)0, sf[j], false, false);
        }

        // Online softmax (scale 1/sqrt(64) = 0.125); DPP16 stays in-half.
#pragma unroll
        for (int r = 0; r < 8; ++r) {
            const float s0 = sf[0][r] * 0.125f;
            const float s1 = sf[1][r] * 0.125f;
            const float cm = row_max16(fmaxf(s0, s1));
            const float nm = fmaxf(rmax[r], cm);
            const float c  = __expf(rmax[r] - nm);
            rmax[r] = nm;
            const float p0 = __expf(s0 - nm);
            const float p1 = __expf(s1 - nm);
            sf[0][r] = p0; sf[1][r] = p1;
            rsum[r] = rsum[r] * c + row_sum16(p0 + p1);
            corr[r] = c;
        }
#pragma unroll
        for (int j2 = 0; j2 < 4; ++j2)
#pragma unroll
            for (int r = 0; r < 8; ++r) o[j2][r] *= corr[r];

        // Re-layout P (f32 D-fragment) -> bf16 A-fragment via per-wave LDS tile.
        __builtin_amdgcn_wave_barrier();
#pragma unroll
        for (int j = 0; j < 2; ++j)
#pragma unroll
            for (int r = 0; r < 8; ++r)
                pbuf[wave][r + 8 * half][j * 16 + nl] = (__bf16)sf[j][r];
        __builtin_amdgcn_wave_barrier();
        const v16bf pa = cat8(*(const v8bf*)&pbuf[wave][nl][half * 8],
                              *(const v8bf*)&pbuf[wave][nl][16 + half * 8]);
        __builtin_amdgcn_wave_barrier();

        // O += P * V  (V tile already [d][key] in LDS, contiguous in key).
#pragma unroll
        for (int j2 = 0; j2 < 4; ++j2) {
            const v16bf vb = cat8(*(const v8bf*)(vlds + (j2 * 16 + nl) * VROW + half * 8),
                                  *(const v8bf*)(vlds + (j2 * 16 + nl) * VROW + 16 + half * 8));
            o[j2] = __builtin_amdgcn_wmma_f32_16x16x32_bf16(false, pa, false, vb, (short)0, o[j2], false, false);
        }

        __syncthreads();   // tile fully consumed before TDM overwrites it
    }

#pragma unroll
    for (int r = 0; r < 8; ++r) rsum[r] = 1.0f / rsum[r];
#pragma unroll
    for (int j2 = 0; j2 < 4; ++j2)
#pragma unroll
        for (int r = 0; r < 8; ++r) {
            const size_t idx = ((size_t)(b * S + q0 + r + 8 * half)) * EMBED
                             + h * HEAD_DIM + j2 * 16 + nl;
            O[idx] = (__bf16)(o[j2][r] * rsum[r]);
        }
}

// ---------------------------------------------------------------------------
extern "C" void kernel_launch(void* const* d_in, const int* in_sizes, int n_in,
                              void* d_out, int out_size, void* d_ws, size_t ws_size,
                              hipStream_t stream) {
    const float* q  = (const float*)d_in[0];
    const float* k  = (const float*)d_in[1];
    const float* v  = (const float*)d_in[2];
    const float* Wq = (const float*)d_in[3];
    const float* bq = (const float*)d_in[4];
    const float* Wk = (const float*)d_in[5];
    const float* bk = (const float*)d_in[6];
    const float* Wv = (const float*)d_in[7];
    const float* bv = (const float*)d_in[8];
    const float* Wo = (const float*)d_in[9];
    const float* bo = (const float*)d_in[10];

    const int S = SEQ, E = EMBED, M = BATCH * SEQ;
    const size_t nME = (size_t)M * E;   // 8M elements
    const size_t nEE = (size_t)E * E;   // 1M elements

    // Workspace (bf16 elements): Xq|Xk|Xv|Wqb|Wkb|Wvb|Wob|Qb|Kb|Vt  (~100 MB).
    // Ob aliases Xq (dead after the Q projection finishes; stream-ordered).
    unsigned short* Xq  = (unsigned short*)d_ws;
    unsigned short* Xk  = Xq  + nME;
    unsigned short* Xv  = Xk  + nME;
    unsigned short* Wqb = Xv  + nME;
    unsigned short* Wkb = Wqb + nEE;
    unsigned short* Wvb = Wkb + nEE;
    unsigned short* Wob = Wvb + nEE;
    unsigned short* Qb  = Wob + nEE;
    unsigned short* Kb  = Qb  + nME;
    unsigned short* Vt  = Kb  + nME;
    unsigned short* Ob  = Xq;           // alias

    // 1) one-shot conversions to bf16
    {
        const int nA8 = (int)(nME / 8), nW8 = (int)(nEE / 8);
        const dim3 cb(256);
        cvt_bf16_kernel<<<dim3((nA8 + 255) / 256), cb, 0, stream>>>(q,  Xq,  nA8);
        cvt_bf16_kernel<<<dim3((nA8 + 255) / 256), cb, 0, stream>>>(k,  Xk,  nA8);
        cvt_bf16_kernel<<<dim3((nA8 + 255) / 256), cb, 0, stream>>>(v,  Xv,  nA8);
        cvt_bf16_kernel<<<dim3((nW8 + 255) / 256), cb, 0, stream>>>(Wq, Wqb, nW8);
        cvt_bf16_kernel<<<dim3((nW8 + 255) / 256), cb, 0, stream>>>(Wk, Wkb, nW8);
        cvt_bf16_kernel<<<dim3((nW8 + 255) / 256), cb, 0, stream>>>(Wv, Wvb, nW8);
        cvt_bf16_kernel<<<dim3((nW8 + 255) / 256), cb, 0, stream>>>(Wo, Wob, nW8);
    }

    const dim3 blk(128);
    const dim3 ggrid(E / 128, M / 128);

    // 2) projections
    gemm_bias_kernel<0><<<ggrid, blk, 0, stream>>>(Xq, Wqb, bq, (void*)Qb, M, E, E, S);
    gemm_bias_kernel<0><<<ggrid, blk, 0, stream>>>(Xk, Wkb, bk, (void*)Kb, M, E, E, S);
    gemm_bias_kernel<2><<<ggrid, blk, 0, stream>>>(Xv, Wvb, bv, (void*)Vt, M, E, E, S);

    // 3) attention (TDM-fed)
    attention_kernel<<<dim3(BATCH * HEADS, S / 64), blk, 0, stream>>>(Qb, Kb, Vt, Ob, S);

    // 4) output projection (f32 out)
    gemm_bias_kernel<1><<<ggrid, blk, 0, stream>>>(Ob, Wob, bo, d_out, M, E, E, S);
}
